// LorentzGNN_764504178734
// MI455X (gfx1250) — compile-verified, matrix operand order checked by reading, not run on hardware
//
#include <hip/hip_runtime.h>
#include <math.h>

// CDNA5 / gfx1250: wave32, WMMA 16x16x32 bf16 -> f32
typedef __attribute__((ext_vector_type(16))) __bf16 v16bf;
typedef __attribute__((ext_vector_type(8)))  float  v8f;

#define N_ROWS 16384
#define D_DIM  64
#define MINN   1e-15f

// ---------------------------------------------------------------------------
// Stage 1: per-row  ht = logmap0(proj(expmap0(logmap0(x) @ W^T)))
// Output written as bf16 in WMMA-B-fragment-friendly packed layout:
//   idx(k,n) = ((kb*64 + n)*2 + half)*16 + elem
//   kb = k/32, kr = k%32, half = bit3(kr), elem = (kr&7) | (bit4(kr)<<3)
// which matches the ISA 16-bit B (32x16) VGPR layout: lane = half*16 + (n%16),
// element e<8  -> K = kb*32 + 8*half + e
// element e>=8 -> K = kb*32 + 16 + 8*half + (e-8)
// ---------------------------------------------------------------------------
__global__ __launch_bounds__(256) void lgnn_stage1(
    const float* __restrict__ x, const float* __restrict__ W,
    unsigned short* __restrict__ bp_raw)
{
  __bf16* Bp = reinterpret_cast<__bf16*>(bp_raw);
  const int i = blockIdx.x * blockDim.x + threadIdx.x;   // row index == K index
  const float* xr = x + (size_t)i * D_DIM;

  // u = logmap0(x): u[0]=0, u[j] = acosh(max(x0,1+1e-7)) * x[j] / max(||x[1:]||,MINN)
  float u[D_DIM];
  float x0 = xr[0];
  float sq = 0.f;
#pragma unroll
  for (int j = 1; j < D_DIM; ++j) { float t = xr[j]; u[j] = t; sq = fmaf(t, t, sq); }
  float yn  = fmaxf(sqrtf(sq), MINN);
  float th  = fmaxf(x0, 1.0f + 1e-7f);
  float fac = acoshf(th) / yn;
  u[0] = 0.f;
#pragma unroll
  for (int j = 1; j < D_DIM; ++j) u[j] *= fac;

  // pass 1: vsq = sum_{n>=1} v_n^2, where v_n = sum_d u[d]*W[n][d]
  // (W indices are wave-uniform -> scalar loads)
  float vsq = 0.f;
  for (int n = 1; n < D_DIM; ++n) {
    const float* wr = W + n * D_DIM;
    float v = 0.f;
#pragma unroll
    for (int d = 0; d < D_DIM; ++d) v = fmaf(u[d], wr[d], v);
    vsq = fmaf(v, v, vsq);
  }

  // h = proj(expmap0(v));  ht = logmap0(h)  => ht[n] = fac2 * v_n (n>=1), ht[0]=0
  float xn = fmaxf(sqrtf(vsq), MINN);
  float sh = sinhf(xn);
  float sumrest = (sh * sh) * (vsq / (xn * xn));      // sum of (sinh*v/xn)^2
  float h0  = sqrtf(fmaxf(1.0f + sumrest, MINN));     // proj first component
  float ynh = fmaxf(fabsf(sh) * sqrtf(vsq) / xn, MINN); // ||h[1:]||
  float th2 = fmaxf(h0, 1.0f + 1e-7f);
  float fac2 = (acoshf(th2) / ynh) * (sh / xn);

  // pack bf16 into WMMA B layout
  const int kb   = i >> 5;
  const int kr   = i & 31;
  const int half = (kr >> 3) & 1;
  const int elem = (kr & 7) | ((kr >> 4) << 3);

  Bp[((size_t)(kb * 64 + 0) * 2 + half) * 16 + elem] = (__bf16)0.f;
  for (int n = 1; n < D_DIM; ++n) {
    const float* wr = W + n * D_DIM;
    float v = 0.f;
#pragma unroll
    for (int d = 0; d < D_DIM; ++d) v = fmaf(u[d], wr[d], v);
    Bp[((size_t)(kb * 64 + n) * 2 + half) * 16 + elem] = (__bf16)(fac2 * v);
  }
}

// ---------------------------------------------------------------------------
// Stage 2: z = adj @ ht (bf16 WMMA, f32 accum), fused rowwise epilogue:
//   out = proj(expmap0(relu(logmap0(proj(expmap0(z))))))
// Block = 32 output rows x all 64 cols. 8 waves: rtile(2) x ct(4), each wave
// owns one 16x16 tile accumulated over K=16384 in steps of 32.
// A fragment (ISA 16-bit A 16x32 layout): lane L (M = L%16, half = L/16):
//   elems 0..7  = adj[row][kb*32 + 8*half + 0..7]       (contiguous f32 x8)
//   elems 8..15 = adj[row][kb*32 + 16 + 8*half + 0..7]  (contiguous f32 x8)
// ---------------------------------------------------------------------------
__global__ __launch_bounds__(256) void lgnn_stage2(
    const float* __restrict__ adj, const unsigned short* __restrict__ bp_raw,
    float* __restrict__ out)
{
  const v16bf* Bp = reinterpret_cast<const v16bf*>(bp_raw);
  __shared__ float zbuf[32][65];                       // +1 pad: bank-conflict-free

  const int tid   = threadIdx.x;
  const int lane  = tid & 31;
  const int wave  = tid >> 5;
  const int rtile = wave >> 2;       // 0..1
  const int ct    = wave & 3;        // 0..3
  const int half  = lane >> 4;       // 0..1
  const int l16   = lane & 15;
  const int rowBase = blockIdx.x * 32;

  const float* arow = adj + (size_t)(rowBase + rtile * 16 + l16) * N_ROWS + half * 8;
  const v16bf* bcol = Bp + ((size_t)(ct * 16 + l16) * 2 + half);

  v8f acc = {};
#pragma unroll 2
  for (int kb = 0; kb < N_ROWS / 32; ++kb) {
    const float* ap = arow + (size_t)kb * 32;
    float4 a0 = *reinterpret_cast<const float4*>(ap + 0);
    float4 a1 = *reinterpret_cast<const float4*>(ap + 4);
    float4 a2 = *reinterpret_cast<const float4*>(ap + 16);
    float4 a3 = *reinterpret_cast<const float4*>(ap + 20);
    v16bf a;
    a[0]  = (__bf16)a0.x; a[1]  = (__bf16)a0.y; a[2]  = (__bf16)a0.z; a[3]  = (__bf16)a0.w;
    a[4]  = (__bf16)a1.x; a[5]  = (__bf16)a1.y; a[6]  = (__bf16)a1.z; a[7]  = (__bf16)a1.w;
    a[8]  = (__bf16)a2.x; a[9]  = (__bf16)a2.y; a[10] = (__bf16)a2.z; a[11] = (__bf16)a2.w;
    a[12] = (__bf16)a3.x; a[13] = (__bf16)a3.y; a[14] = (__bf16)a3.z; a[15] = (__bf16)a3.w;

    v16bf b = bcol[(size_t)kb * 128];   // fragment stride per K-step = 64*2 v16bf

    acc = __builtin_amdgcn_wmma_f32_16x16x32_bf16(
        /*neg_a=*/false, a, /*neg_b=*/false, b,
        /*c_mod=*/(short)0, acc, /*reuse_a=*/false, /*reuse_b=*/false);
  }

  // D layout: VGPR r -> M = rtile*16 + 8*half + r ; N = ct*16 + l16
  const int n = ct * 16 + l16;
#pragma unroll
  for (int r = 0; r < 8; ++r) {
    zbuf[rtile * 16 + 8 * half + r][n] = acc[r];
  }
  __syncthreads();

  // Rowwise epilogue: wave 0, one row per lane (32 rows per block)
  if (tid < 32) {
    float z[D_DIM];
#pragma unroll
    for (int j = 0; j < D_DIM; ++j) z[j] = zbuf[tid][j];

    float sq = 0.f;
#pragma unroll
    for (int j = 1; j < D_DIM; ++j) sq = fmaf(z[j], z[j], sq);
    float xn  = fmaxf(sqrtf(sq), MINN);
    float sh  = sinhf(xn);
    float sumrest = (sh * sh) * (sq / (xn * xn));
    float h0  = sqrtf(fmaxf(1.0f + sumrest, MINN));
    float ynh = fmaxf(fabsf(sh) * sqrtf(sq) / xn, MINN);
    float th2 = fmaxf(h0, 1.0f + 1e-7f);
    float fac2 = (acoshf(th2) / ynh) * (sh / xn);   // logmap0 of proj(expmap0(z))

    float xt[D_DIM];
    float sq2 = 0.f;
#pragma unroll
    for (int j = 1; j < D_DIM; ++j) {
      float t = fmaxf(fac2 * z[j], 0.f);            // relu (col 0 is already 0)
      xt[j] = t;
      sq2 = fmaf(t, t, sq2);
    }
    float xn2 = fmaxf(sqrtf(sq2), MINN);
    float sh2 = sinhf(xn2);
    float fo  = sh2 / xn2;
    float s2  = (sh2 * sh2) * (sq2 / (xn2 * xn2));
    float o0  = sqrtf(fmaxf(1.0f + s2, MINN));      // proj(expmap0(xt)) first comp

    float* orow = out + (size_t)(rowBase + tid) * D_DIM;
    orow[0] = o0;
#pragma unroll
    for (int j = 1; j < D_DIM; ++j) orow[j] = fo * xt[j];
  }
}

// ---------------------------------------------------------------------------
extern "C" void kernel_launch(void* const* d_in, const int* in_sizes, int n_in,
                              void* d_out, int out_size, void* d_ws, size_t ws_size,
                              hipStream_t stream) {
  const float* x   = (const float*)d_in[0];   // (16384, 64)
  const float* adj = (const float*)d_in[1];   // (16384, 16384)
  const float* W   = (const float*)d_in[2];   // (64, 64)
  float* out = (float*)d_out;                 // (16384, 64)

  // workspace: packed bf16 ht, 16384*64*2 B = 2 MB (L2-resident during stage 2)
  unsigned short* Bp = (unsigned short*)d_ws;

  lgnn_stage1<<<N_ROWS / 256, 256, 0, stream>>>(x, W, Bp);
  lgnn_stage2<<<N_ROWS / 32, 256, 0, stream>>>(adj, Bp, out);
}
